// CustomGAT_46626164965921
// MI455X (gfx1250) — compile-verified, matrix operand order checked by reading, not run on hardware
//
#include <hip/hip_runtime.h>

typedef _Float16 v16h __attribute__((ext_vector_type(16)));
typedef _Float16 v8h  __attribute__((ext_vector_type(8)));
typedef _Float16 v4h  __attribute__((ext_vector_type(4)));
typedef float    v8f  __attribute__((ext_vector_type(8)));

#define NEG_SLOPE 0.2f
#define LN_EPS    1e-5f
#define LDSTR     136   // halves per LDS row (128 + 8 pad -> conflict-free tile loads)

// ---- monotonic float<->uint encoding so unsigned atomicMax == float max ----
__device__ __forceinline__ unsigned fenc(float f) {
  unsigned u = __float_as_uint(f);
  return (u & 0x80000000u) ? ~u : (u | 0x80000000u);
}
__device__ __forceinline__ float fdec(unsigned u) {
  return __uint_as_float((u & 0x80000000u) ? (u & 0x7fffffffu) : ~u);
}

// ---------------- utility: zero a float4 region ----------------
__global__ __launch_bounds__(256) void k_zero4(float4* __restrict__ p, int n4) {
  int i = blockIdx.x * 256 + threadIdx.x;
  if (i < n4) p[i] = make_float4(0.f, 0.f, 0.f, 0.f);
}

// ---------------- embedding gather: x[n,:] = emb[tok[n],:] ----------------
__global__ __launch_bounds__(256) void k_embed(const int* __restrict__ tok,
                                               const float* __restrict__ emb,
                                               float* __restrict__ x, int N) {
  int id = blockIdx.x * 256 + threadIdx.x;
  int n = id >> 5, c = id & 31;
  if (n < N) {
    int t = tok[n];
    float4 v = *(const float4*)&emb[(size_t)t * 128 + c * 4];
    *(float4*)&x[(size_t)n * 128 + c * 4] = v;
  }
}

// ---------------- Wt[n][k] = (f16) W[k][n] ----------------
__global__ __launch_bounds__(256) void k_wt(const float* __restrict__ W,
                                            _Float16* __restrict__ Wt) {
  int id = blockIdx.x * 256 + threadIdx.x;  // 16384 total
  int n = id >> 7, k = id & 127;
  Wt[n * 128 + k] = (_Float16)W[k * 128 + n];
}

// ---------------- fused LayerNorm + GEMM (h = LN(x) @ W) via WMMA ----------------
// Block: 64 rows x 128 cols, 256 threads (8 waves). K = 128 in 4 chunks of 32.
__global__ __launch_bounds__(256) void k_ln_gemm(const float* __restrict__ x,
                                                 const float* __restrict__ ln_s,
                                                 const float* __restrict__ ln_b,
                                                 const _Float16* __restrict__ Wt,
                                                 float* __restrict__ h) {
  __shared__ _Float16 xl[64 * LDSTR];
  __shared__ _Float16 wl[128 * LDSTR];
  const int tid  = threadIdx.x;
  const int wave = tid >> 5;
  const int lane = tid & 31;
  const int row_base = blockIdx.x * 64;

  // stage Wt (n-major, 128x128 f16) into LDS
  for (int i = tid; i < 2048; i += 256) {
    int r = i >> 4, c8 = i & 15;
    *(v8h*)&wl[r * LDSTR + c8 * 8] = *(const v8h*)&Wt[r * 128 + c8 * 8];
  }

  // LayerNorm 8 rows per wave -> f16 in LDS
  for (int i = 0; i < 8; ++i) {
    int row = wave * 8 + i;
    float4 v = *(const float4*)&x[(size_t)(row_base + row) * 128 + lane * 4];
    float sum = v.x + v.y + v.z + v.w;
    float sq  = v.x * v.x + v.y * v.y + v.z * v.z + v.w * v.w;
    for (int m = 16; m >= 1; m >>= 1) {
      sum += __shfl_xor(sum, m);
      sq  += __shfl_xor(sq, m);
    }
    float mu  = sum * (1.f / 128.f);
    float var = sq * (1.f / 128.f) - mu * mu;
    float inv = rsqrtf(var + LN_EPS);
    float4 sc = *(const float4*)&ln_s[lane * 4];
    float4 bi = *(const float4*)&ln_b[lane * 4];
    v4h o;
    o[0] = (_Float16)((v.x - mu) * inv * sc.x + bi.x);
    o[1] = (_Float16)((v.y - mu) * inv * sc.y + bi.y);
    o[2] = (_Float16)((v.z - mu) * inv * sc.z + bi.z);
    o[3] = (_Float16)((v.w - mu) * inv * sc.w + bi.w);
    *(v4h*)&xl[row * LDSTR + lane * 4] = o;
  }
  __syncthreads();

  // WMMA: wave -> row-tile rt (16 rows), 4 column tiles
  const int rt   = wave & 3;
  const int cg   = wave >> 2;
  const int l16  = lane & 15;
  const int sub  = lane >> 4;           // half-wave K-split per ISA layout
  const int mrow = rt * 16 + l16;

  v8f acc0 = {0.f,0.f,0.f,0.f,0.f,0.f,0.f,0.f};
  v8f acc1 = acc0, acc2 = acc0, acc3 = acc0;
  v8f acc[4] = {acc0, acc1, acc2, acc3};

#pragma unroll
  for (int kc = 0; kc < 4; ++kc) {
    const int k0 = kc * 32;
    v8h alo = *(const v8h*)&xl[mrow * LDSTR + k0 + sub * 8];
    v8h ahi = *(const v8h*)&xl[mrow * LDSTR + k0 + 16 + sub * 8];
    v16h A;
#pragma unroll
    for (int q = 0; q < 8; ++q) { A[q] = alo[q]; A[8 + q] = ahi[q]; }
#pragma unroll
    for (int t = 0; t < 4; ++t) {
      const int n = (cg * 4 + t) * 16 + l16;
      v8h blo = *(const v8h*)&wl[n * LDSTR + k0 + sub * 8];
      v8h bhi = *(const v8h*)&wl[n * LDSTR + k0 + 16 + sub * 8];
      v16h B;
#pragma unroll
      for (int q = 0; q < 8; ++q) { B[q] = blo[q]; B[8 + q] = bhi[q]; }
      acc[t] = __builtin_amdgcn_wmma_f32_16x16x32_f16(
          false, A, false, B, (short)0, acc[t], false, false);
    }
  }

#pragma unroll
  for (int t = 0; t < 4; ++t) {
    const int col = (cg * 4 + t) * 16 + l16;
#pragma unroll
    for (int v = 0; v < 8; ++v) {
      int M = rt * 16 + v + sub * 8;    // C/D layout: M = vgpr + 8*(lane>=16)
      h[(size_t)(row_base + M) * 128 + col] = acc[t][v];
    }
  }
}

// ---------------- per-node scores: ss = h@a_src, sd = h@a_dst ----------------
__global__ __launch_bounds__(256) void k_scores(const float* __restrict__ h,
                                                const float* __restrict__ as_,
                                                const float* __restrict__ ad_,
                                                float* __restrict__ ss,
                                                float* __restrict__ sd, int N) {
  int wave = threadIdx.x >> 5, lane = threadIdx.x & 31;
  int n = blockIdx.x * 8 + wave;
  if (n >= N) return;
  float4 v = *(const float4*)&h[(size_t)n * 128 + lane * 4];
  float4 a = *(const float4*)&as_[lane * 4];
  float4 b = *(const float4*)&ad_[lane * 4];
  float ds_ = v.x * a.x + v.y * a.y + v.z * a.z + v.w * a.w;
  float dd_ = v.x * b.x + v.y * b.y + v.z * b.z + v.w * b.w;
  for (int m = 16; m >= 1; m >>= 1) {
    ds_ += __shfl_xor(ds_, m);
    dd_ += __shfl_xor(dd_, m);
  }
  if (lane == 0) { ss[n] = ds_; sd[n] = dd_; }
}

// ---------------- pass 1: alpha + segment max ----------------
__global__ __launch_bounds__(256) void k_alpha_max(const int* __restrict__ ei,
                                                   const float* __restrict__ ea,
                                                   const float* __restrict__ ss,
                                                   const float* __restrict__ sd,
                                                   float* __restrict__ albuf,
                                                   unsigned* __restrict__ menc,
                                                   int E, int Etot) {
  int e = blockIdx.x * 256 + threadIdx.x;
  if (e >= Etot) return;
  int s_, d_; float w;
  if (e < E) { s_ = ei[e]; d_ = ei[E + e]; w = ea[e]; }
  else       { s_ = d_ = e - E; w = 1.f; }
  float al = sd[d_] + ss[s_];
  al = (al >= 0.f) ? al : NEG_SLOPE * al;
  al *= w;
  albuf[e] = al;
  atomicMax(&menc[d_], fenc(al));
}

// ---------------- pass 2: exp + segment sum ----------------
__global__ __launch_bounds__(256) void k_expsum(const int* __restrict__ ei,
                                                const float* __restrict__ albuf_in,
                                                float* __restrict__ albuf,
                                                const unsigned* __restrict__ menc,
                                                float* __restrict__ sbuf,
                                                int E, int Etot) {
  int e = blockIdx.x * 256 + threadIdx.x;
  if (e >= Etot) return;
  int d_ = (e < E) ? ei[E + e] : (e - E);
  float ex = __expf(albuf_in[e] - fdec(menc[d_]));
  albuf[e] = ex;
  atomicAdd(&sbuf[d_], ex);
}

// ---------------- pass 3: normalized scatter-add aggregation (wave/edge) ------
__global__ __launch_bounds__(256) void k_aggregate(const int* __restrict__ ei,
                                                   const float* __restrict__ albuf,
                                                   const float* __restrict__ sbuf,
                                                   const float* __restrict__ h,
                                                   float* __restrict__ agg,
                                                   int E, int Etot) {
  int wave = threadIdx.x >> 5, lane = threadIdx.x & 31;
  int e = blockIdx.x * 8 + wave;
  if (e >= Etot) return;
  int s_, d_;
  if (e < E) { s_ = ei[e]; d_ = ei[E + e]; }
  else       { s_ = d_ = e - E; }
  float a = albuf[e] / sbuf[d_];
  float4 v = *(const float4*)&h[(size_t)s_ * 128 + lane * 4];
  float* o = &agg[(size_t)d_ * 128 + lane * 4];
  atomicAdd(o + 0, v.x * a);
  atomicAdd(o + 1, v.y * a);
  atomicAdd(o + 2, v.z * a);
  atomicAdd(o + 3, v.w * a);
}

// ---------------- elementwise relu ----------------
__global__ __launch_bounds__(256) void k_relu(float4* __restrict__ p, int n4) {
  int i = blockIdx.x * 256 + threadIdx.x;
  if (i >= n4) return;
  float4 v = p[i];
  v.x = v.x > 0.f ? v.x : 0.f;
  v.y = v.y > 0.f ? v.y : 0.f;
  v.z = v.z > 0.f ? v.z : 0.f;
  v.w = v.w > 0.f ? v.w : 0.f;
  p[i] = v;
}

// ---------------- global mean-pool accumulation ----------------
__global__ __launch_bounds__(256) void k_pool(const float* __restrict__ x,
                                              const int* __restrict__ batch,
                                              float* __restrict__ gsum,
                                              float* __restrict__ cnt, int N) {
  int wave = threadIdx.x >> 5, lane = threadIdx.x & 31;
  int n = blockIdx.x * 8 + wave;
  if (n >= N) return;
  int b = batch[n];
  float4 v = *(const float4*)&x[(size_t)n * 128 + lane * 4];
  float* o = &gsum[(size_t)b * 128 + lane * 4];
  atomicAdd(o + 0, v.x);
  atomicAdd(o + 1, v.y);
  atomicAdd(o + 2, v.z);
  atomicAdd(o + 3, v.w);
  if (lane == 0) atomicAdd(&cnt[b], 1.f);
}

// ---------------- final MLP: 128 -> 64 (relu) -> 2 ----------------
__global__ __launch_bounds__(64) void k_mlp(const float* __restrict__ gsum,
                                            const float* __restrict__ cnt,
                                            const float* __restrict__ w1,
                                            const float* __restrict__ b1,
                                            const float* __restrict__ w2,
                                            const float* __restrict__ b2,
                                            float* __restrict__ out) {
  __shared__ float g[128];
  __shared__ float h1[64];
  int b = blockIdx.x, t = threadIdx.x;
  float c = cnt[b]; c = c > 1.f ? c : 1.f;
  g[t]      = gsum[b * 128 + t] / c;
  g[t + 64] = gsum[b * 128 + 64 + t] / c;
  __syncthreads();
  float acc = b1[t];
  for (int k = 0; k < 128; ++k) acc += g[k] * w1[k * 64 + t];
  h1[t] = acc > 0.f ? acc : 0.f;
  __syncthreads();
  if (t < 2) {
    float o = b2[t];
    for (int j = 0; j < 64; ++j) o += h1[j] * w2[j * 2 + t];
    out[b * 2 + t] = o;
  }
}

extern "C" void kernel_launch(void* const* d_in, const int* in_sizes, int n_in,
                              void* d_out, int out_size, void* d_ws, size_t ws_size,
                              hipStream_t stream) {
  const int*   tok   = (const int*)d_in[0];
  const int*   ei    = (const int*)d_in[1];
  const float* ea    = (const float*)d_in[2];
  const int*   batch = (const int*)d_in[3];
  const float* emb   = (const float*)d_in[4];
  const float* lns[2] = {(const float*)d_in[5],  (const float*)d_in[10]};
  const float* lnb[2] = {(const float*)d_in[6],  (const float*)d_in[11]};
  const float* W[2]   = {(const float*)d_in[7],  (const float*)d_in[12]};
  const float* as_[2] = {(const float*)d_in[8],  (const float*)d_in[13]};
  const float* ad_[2] = {(const float*)d_in[9],  (const float*)d_in[14]};
  const float* w1 = (const float*)d_in[15];
  const float* b1 = (const float*)d_in[16];
  const float* w2 = (const float*)d_in[17];
  const float* b2 = (const float*)d_in[18];
  float* out = (float*)d_out;

  const int N    = in_sizes[0];
  const int E    = in_sizes[1] / 2;
  const int Etot = E + N;
  const int NP   = ((N + 63) / 64) * 64;   // rows padded so WMMA tiles need no predication

  auto alignup = [](size_t x) { return (x + 255) & ~(size_t)255; };
  char* ws = (char*)d_ws;
  size_t o = 0;
  float*     bufA  = (float*)(ws + o);     o = alignup(o + (size_t)NP * 128 * 4);
  float*     bufB  = (float*)(ws + o);     o = alignup(o + (size_t)NP * 128 * 4);
  float*     albuf = (float*)(ws + o);     o = alignup(o + (size_t)Etot * 4);
  float*     ssb   = (float*)(ws + o);     o = alignup(o + (size_t)NP * 4);
  float*     sdb   = (float*)(ws + o);     o = alignup(o + (size_t)NP * 4);
  unsigned*  menc  = (unsigned*)(ws + o);  o = alignup(o + (size_t)NP * 4);
  float*     sbuf  = (float*)(ws + o);     o = alignup(o + (size_t)NP * 4);
  _Float16*  Wt    = (_Float16*)(ws + o);  o = alignup(o + 128 * 128 * 2);
  float*     gsum  = (float*)(ws + o);     o = alignup(o + 128 * 128 * 4);
  float*     cnt   = (float*)(ws + o);     o = alignup(o + 128 * 4);

  const int featN4 = NP * 32;              // NP*128/4

  // x <- 0 (incl. pad rows), then embedding gather
  k_zero4<<<(featN4 + 255) / 256, 256, 0, stream>>>((float4*)bufA, featN4);
  k_embed<<<(N * 32 + 255) / 256, 256, 0, stream>>>(tok, emb, bufA, N);

  for (int l = 0; l < 2; ++l) {
    // h = LN(x) @ W  (WMMA f16 -> f32)
    k_wt<<<64, 256, 0, stream>>>(W[l], Wt);
    k_ln_gemm<<<NP / 64, 256, 0, stream>>>(bufA, lns[l], lnb[l], Wt, bufB);
    // per-node attention scores
    k_scores<<<(N + 7) / 8, 256, 0, stream>>>(bufB, as_[l], ad_[l], ssb, sdb, N);
    // segment softmax over dst (3 passes via L2 atomics)
    k_zero4<<<(NP / 4 + 255) / 256, 256, 0, stream>>>((float4*)menc, NP / 4);
    k_zero4<<<(NP / 4 + 255) / 256, 256, 0, stream>>>((float4*)sbuf, NP / 4);
    k_alpha_max<<<(Etot + 255) / 256, 256, 0, stream>>>(ei, ea, ssb, sdb, albuf, menc, E, Etot);
    k_expsum<<<(Etot + 255) / 256, 256, 0, stream>>>(ei, albuf, albuf, menc, sbuf, E, Etot);
    // aggregate weighted h[src] into x-buffer (L2-resident scatter-add)
    k_zero4<<<(featN4 + 255) / 256, 256, 0, stream>>>((float4*)bufA, featN4);
    k_aggregate<<<(Etot + 7) / 8, 256, 0, stream>>>(ei, albuf, sbuf, bufB, bufA, E, Etot);
    k_relu<<<(N * 32 + 255) / 256, 256, 0, stream>>>((float4*)bufA, N * 32);
  }

  // mean pool + MLP head
  k_zero4<<<(128 * 32 + 255) / 256, 256, 0, stream>>>((float4*)gsum, 128 * 32);
  k_zero4<<<1, 256, 0, stream>>>((float4*)cnt, 32);
  k_pool<<<(N + 7) / 8, 256, 0, stream>>>(bufA, batch, gsum, cnt, N);
  k_mlp<<<128, 64, 0, stream>>>(gsum, cnt, w1, b1, w2, b2, out);
}
